// MLAttention_54348516163814
// MI455X (gfx1250) — matrix operand order repaired and checked by reading, not compiled, not run on hardware
//
#include <hip/hip_runtime.h>
#include <hip/hip_bf16.h>

// ---------------------------------------------------------------------------
// MLA forward for MI455X (gfx1250, wave32, WMMA + Tensor Data Mover).
// f32->bf16 convert (weights transposed), TDM-staged WMMA GEMMs, RoPE
// assembly, TDM-staged flash attention, final WMMA GEMM -> f32 out.
// ---------------------------------------------------------------------------

typedef __attribute__((ext_vector_type(16))) __bf16 v16bf;
typedef __attribute__((ext_vector_type(8)))  float  v8f;
typedef __attribute__((ext_vector_type(4)))  unsigned int uint4v;
typedef __attribute__((ext_vector_type(4)))  unsigned int u32x4;
typedef __attribute__((ext_vector_type(8)))  int i32x8;
typedef __attribute__((ext_vector_type(4)))  int i32x4;

__device__ __forceinline__ unsigned short f2bf(float f) {
    unsigned int u = __float_as_uint(f);
    u = (u + 0x7fffu + ((u >> 16) & 1u)) >> 16;   // round-to-nearest-even
    return (unsigned short)u;
}
__device__ __forceinline__ float bf2f(unsigned short h) {
    return __uint_as_float(((unsigned int)h) << 16);
}

__device__ __forceinline__ v16bf ld_frag2(const unsigned short* p0,
                                          const unsigned short* p1) {
    union { uint4v u[2]; v16bf v; } U;
    U.u[0] = *(const uint4v*)p0;
    U.u[1] = *(const uint4v*)p1;
    return U.v;
}

__device__ __forceinline__ v8f wmma_bf16(v16bf a, v16bf b, v8f c) {
    return __builtin_amdgcn_wmma_f32_16x16x32_bf16(
        false, a, false, b, (short)0, c, false, false);
}

__device__ __forceinline__ void store_c(float* p, float v) { *p = v; }
__device__ __forceinline__ void store_c(unsigned short* p, float v) { *p = f2bf(v); }

// ---------------------------------------------------------------------------
// Tensor Data Mover: 2D tile load Global -> LDS with per-row LDS padding.
// D# layout per CDNA5 ISA 8.3/8.4: group0 = {flags, lds_addr, global_addr,
// type=2}; group1 = {data_size=2B, pad ctl, tensor dims, tile dims, stride}.
// ---------------------------------------------------------------------------
__device__ __forceinline__ void tdm_load_2d(unsigned lds_off, const void* gptr,
                                            unsigned tensor_d0, unsigned tensor_d1,
                                            unsigned tile_d0, unsigned tile_d1,
                                            unsigned stride0,
                                            unsigned pad_interval, unsigned pad_amount) {
    unsigned long long ga = (unsigned long long)(uintptr_t)gptr;
    u32x4 g0;
    g0[0] = 1u;                                           // count=1 (valid user D#)
    g0[1] = lds_off;                                      // LDS byte address
    g0[2] = (unsigned)(ga & 0xffffffffu);                 // global_addr[31:0]
    g0[3] = (unsigned)((ga >> 32) & 0x01ffffffu)          // global_addr[56:32]
          | (2u << 30);                                   // type=2 ("image")
    i32x8 g1;
    g1[0] = (int)((1u << 16)                              // data_size: 2 bytes
          | (1u << 20)                                    // pad_enable
          | (pad_interval << 22) | (pad_amount << 25));
    g1[1] = (int)((tensor_d0 & 0xffffu) << 16);           // tensor_dim0[15:0]
    g1[2] = (int)((tensor_d0 >> 16) | ((tensor_d1 & 0xffffu) << 16));
    g1[3] = (int)((tensor_d1 >> 16) | (tile_d0 << 16));   // tile_dim0
    g1[4] = (int)(tile_d1 & 0xffffu);                     // tile_dim1 (tile_dim2=0)
    g1[5] = (int)stride0;                                 // tensor_dim0_stride[31:0]
    g1[6] = 0;
    g1[7] = 0;
    i32x4 z4 = {0, 0, 0, 0};
#if defined(__clang_major__) && (__clang_major__ >= 23)
    i32x8 z8 = {0, 0, 0, 0, 0, 0, 0, 0};
    __builtin_amdgcn_tensor_load_to_lds(g0, g1, z4, z4, z8, 0);
#else
    __builtin_amdgcn_tensor_load_to_lds(g0, g1, z4, z4, 0);
#endif
}

// ---------------------------------------------------------------------------
// f32 -> bf16 conversions (plain and transposing, for weights / V layout)
// ---------------------------------------------------------------------------
__global__ void cvt_f32_bf16(const float* __restrict__ in,
                             unsigned short* __restrict__ out, long long n) {
    long long i = (long long)blockIdx.x * blockDim.x + threadIdx.x;
    if (i < n) out[i] = f2bf(in[i]);
}

// out[c*R + r] = in[r*C + c]   (write-coalesced transpose + convert)
__global__ void cvt_f32_bf16_t(const float* __restrict__ in,
                               unsigned short* __restrict__ out, int R, int C) {
    long long i = (long long)blockIdx.x * blockDim.x + threadIdx.x;
    if (i >= (long long)R * C) return;
    int r = (int)(i % R);
    int c = (int)(i / R);
    out[(size_t)c * R + r] = f2bf(in[(size_t)r * C + c]);
}

// ---------------------------------------------------------------------------
// Tiled WMMA GEMM: C(MxN) = A(MxK) * Bt(NxK)^T, bf16 in, f32/bf16 out.
// Both operand tiles staged by TDM with padded LDS rows (stride 40 elems).
// ---------------------------------------------------------------------------
#define GBM 128
#define GBN 128
#define GBK 32
#define TPAD 40   // 32 data + 8 pad elems per LDS row (16B-aligned, bank-safe)

template <typename OUT_T>
__global__ __launch_bounds__(256)
void gemm_bf16(const unsigned short* __restrict__ A,
               const unsigned short* __restrict__ Bt,   // N x K row-major
               OUT_T* __restrict__ C,
               int M, int N, int K) {
    __shared__ unsigned short As[GBM * TPAD];
    __shared__ unsigned short Bs[GBN * TPAD];

    const int tid  = threadIdx.x;
    const int lane = tid & 31;
    const int wave = tid >> 5;
    const int wm   = wave >> 2;       // 0..1 : 64 rows
    const int wn   = wave & 3;        // 0..3 : 32 cols
    const int g    = lane >> 4;
    const int ln   = lane & 15;
    const int bm   = blockIdx.y * GBM;
    const int bn   = blockIdx.x * GBN;

    const unsigned lds_as = (unsigned)(size_t)&As[0];
    const unsigned lds_bs = (unsigned)(size_t)&Bs[0];

    v8f acc[4][2] = {};

    for (int k0 = 0; k0 < K; k0 += GBK) {
        __syncthreads();                       // previous tile fully consumed
        if (wave == 0) {
            // 128 rows x 32 elems, 64B rows + 16B LDS pad -> stride 40 elems
            tdm_load_2d(lds_as, A  + (size_t)bm * K + k0, K, M, GBK, GBM, K, 3u, 3u);
            tdm_load_2d(lds_bs, Bt + (size_t)bn * K + k0, K, N, GBK, GBN, K, 3u, 3u);
        }
        __builtin_amdgcn_s_wait_tensorcnt(0);
        __syncthreads();

        v16bf af[4], bfr[2];
        #pragma unroll
        for (int i = 0; i < 4; ++i) {
            int row = wm * 64 + i * 16 + ln;
            const unsigned short* pa = &As[row * TPAD + g * 8];
            af[i] = ld_frag2(pa, pa + 16);
        }
        #pragma unroll
        for (int j = 0; j < 2; ++j) {
            int col = wn * 32 + j * 16 + ln;
            const unsigned short* pb = &Bs[col * TPAD + g * 8];
            bfr[j] = ld_frag2(pb, pb + 16);
        }
        #pragma unroll
        for (int i = 0; i < 4; ++i)
            #pragma unroll
            for (int j = 0; j < 2; ++j)
                acc[i][j] = wmma_bf16(af[i], bfr[j], acc[i][j]);
    }

    #pragma unroll
    for (int i = 0; i < 4; ++i) {
        #pragma unroll
        for (int j = 0; j < 2; ++j) {
            int row0 = bm + wm * 64 + i * 16 + g * 8;
            int col  = bn + wn * 32 + j * 16 + ln;
            #pragma unroll
            for (int r = 0; r < 8; ++r)
                store_c(C + (size_t)(row0 + r) * N + col, acc[i][j][r]);
        }
    }
}

// ---------------------------------------------------------------------------
// Assemble Q: xq = [q_u, rope(q_r)] * (1/sqrt(2*HD)), Q[((b*16+h)*2048+s)*256+d]
// ---------------------------------------------------------------------------
__global__ void assemble_q(const unsigned short* __restrict__ qu,
                           const unsigned short* __restrict__ qr,
                           const float* __restrict__ cosT,
                           const float* __restrict__ sinT,
                           unsigned short* __restrict__ Q) {
    long long idx = (long long)blockIdx.x * blockDim.x + threadIdx.x; // B*S*16*64
    int p = (int)(idx & 63);
    int h = (int)((idx >> 6) & 15);
    int s = (int)((idx >> 10) & 2047);
    int b = (int)(idx >> 21);
    const float scale = 1.0f / 16.0f;                 // 1/sqrt(2*128)
    float c  = cosT[s * 64 + p];
    float sn = sinT[s * 64 + p];
    size_t rin  = (size_t)b * 2048 + s;
    size_t icol = (size_t)h * 128 + 2 * p;
    float u0 = bf2f(qu[rin * 2048 + icol]);
    float u1 = bf2f(qu[rin * 2048 + icol + 1]);
    float r0 = bf2f(qr[rin * 2048 + icol]);
    float r1 = bf2f(qr[rin * 2048 + icol + 1]);
    size_t orow = ((size_t)(b * 16 + h) * 2048 + s) * 256;
    Q[orow + 2 * p]           = f2bf(u0 * scale);
    Q[orow + 2 * p + 1]       = f2bf(u1 * scale);
    Q[orow + 128 + 2 * p]     = f2bf((r0 * c - r1 * sn) * scale);
    Q[orow + 128 + 2 * p + 1] = f2bf((r0 * sn + r1 * c) * scale);
}

// Assemble K: xk = [k_u, rope(k_r)]  ->  K[((b*8+kv)*2048+s)*256+d]
__global__ void assemble_k(const unsigned short* __restrict__ ku,
                           const unsigned short* __restrict__ krp,
                           const float* __restrict__ cosT,
                           const float* __restrict__ sinT,
                           unsigned short* __restrict__ Ko) {
    long long idx = (long long)blockIdx.x * blockDim.x + threadIdx.x; // B*S*8*64
    int p = (int)(idx & 63);
    int h = (int)((idx >> 6) & 7);
    int s = (int)((idx >> 9) & 2047);
    int b = (int)(idx >> 20);
    float c  = cosT[s * 64 + p];
    float sn = sinT[s * 64 + p];
    size_t rin  = (size_t)b * 2048 + s;
    size_t icol = (size_t)h * 128 + 2 * p;
    float u0 = bf2f(ku[rin * 1024 + icol]);
    float u1 = bf2f(ku[rin * 1024 + icol + 1]);
    float r0 = bf2f(krp[rin * 1024 + icol]);
    float r1 = bf2f(krp[rin * 1024 + icol + 1]);
    size_t orow = ((size_t)(b * 8 + h) * 2048 + s) * 256;
    Ko[orow + 2 * p]           = f2bf(u0);
    Ko[orow + 2 * p + 1]       = f2bf(u1);
    Ko[orow + 128 + 2 * p]     = f2bf(r0 * c - r1 * sn);
    Ko[orow + 128 + 2 * p + 1] = f2bf(r0 * sn + r1 * c);
}

// Assemble V transposed: Vt[((b*8+kv)*128 + n)*2048 + s] = v_u[(b*2048+s)*1024 + kv*128+n]
__global__ void assemble_v(const unsigned short* __restrict__ vu,
                           unsigned short* __restrict__ Vt) {
    long long idx = (long long)blockIdx.x * blockDim.x + threadIdx.x; // B*8*128*2048
    int s  = (int)(idx & 2047);
    int n  = (int)((idx >> 11) & 127);
    int kv = (int)((idx >> 18) & 7);
    int b  = (int)(idx >> 21);
    Vt[idx] = vu[((size_t)b * 2048 + s) * 1024 + (size_t)kv * 128 + n];
}

// ---------------------------------------------------------------------------
// Causal flash attention. 4 waves x 16-row Q strips, 32-key tiles staged by
// TDM (K row-major 264-elem rows; V pre-transposed, 40-elem rows).
// ---------------------------------------------------------------------------
#define AQT 64
#define AKT 32
#define KSPAD 264   // 256 + 8 pad elems (TDM pad: 512B interval, 16B pad)
#define VTPAD 40
#define PSPAD 40

__global__ __launch_bounds__(128)
void mla_attention(const unsigned short* __restrict__ Q,
                   const unsigned short* __restrict__ K,
                   const unsigned short* __restrict__ Vt,    // (B*8*128) x 2048
                   unsigned short* __restrict__ O) {         // (4096 x 2048)
    __shared__ unsigned short Ks[AKT * KSPAD];
    __shared__ unsigned short Vs[128 * VTPAD];
    __shared__ unsigned short Ps[4 * 16 * PSPAD];

    const int qb   = blockIdx.x * AQT;
    const int h    = blockIdx.y;
    const int b    = blockIdx.z;
    const int kvh  = h >> 1;                                  // N_REP = 2
    const int tid  = threadIdx.x;
    const int lane = tid & 31;
    const int wave = tid >> 5;
    const int g    = lane >> 4;
    const int ln   = lane & 15;

    const unsigned short* Qb = Q  + (size_t)(b * 16 + h) * 2048 * 256;
    const unsigned short* Kb = K  + (size_t)(b * 8 + kvh) * 2048 * 256;
    const unsigned short* Vb = Vt + (size_t)(b * 8 + kvh) * 128 * 2048;

    const unsigned lds_ks = (unsigned)(size_t)&Ks[0];
    const unsigned lds_vs = (unsigned)(size_t)&Vs[0];

    // Resident Q strip: 16 rows x 256 -> 8 A-fragments.
    const int qrow = qb + wave * 16 + ln;
    v16bf qf[8];
    #pragma unroll
    for (int c = 0; c < 8; ++c) {
        const unsigned short* p = Qb + (size_t)qrow * 256 + c * 32 + g * 8;
        qf[c] = ld_frag2(p, p + 16);
    }

    v8f o[8] = {};
    float m[8], lsum[8];
    #pragma unroll
    for (int r = 0; r < 8; ++r) { m[r] = -1e30f; lsum[r] = 0.0f; }

    const int qmax = qb + wave * 16 + 15;
    const int kend = qb + AQT;

    for (int kb = 0; kb < kend; kb += AKT) {
        __syncthreads();
        if (wave == 0) {
            // K tile: 32 rows x 256 elems (512B) + 16B pad -> 264-elem rows
            tdm_load_2d(lds_ks, Kb + (size_t)kb * 256, 256u, 2048u,
                        256u, AKT, 256u, 6u, 3u);
            // V tile: 128 rows x 32 elems (64B) + 16B pad -> 40-elem rows
            tdm_load_2d(lds_vs, Vb + kb, 2048u, 128u,
                        AKT, 128u, 2048u, 3u, 3u);
        }
        __builtin_amdgcn_s_wait_tensorcnt(0);
        __syncthreads();

        if (kb > qmax) continue;   // fully-masked tile for this wave's strip

        // ---- scores: S(16x32) = Q(16x256) . K^T, two 16x16 WMMA chains ----
        v8f s0 = {}, s1 = {};
        #pragma unroll
        for (int c = 0; c < 8; ++c) {
            const unsigned short* pk0 = &Ks[ln * KSPAD + c * 32 + g * 8];
            const unsigned short* pk1 = &Ks[(16 + ln) * KSPAD + c * 32 + g * 8];
            v16bf kf0 = ld_frag2(pk0, pk0 + 16);
            v16bf kf1 = ld_frag2(pk1, pk1 + 16);
            s0 = wmma_bf16(qf[c], kf0, s0);
            s1 = wmma_bf16(qf[c], kf1, s1);
        }

        // ---- causal mask (C-layout: row = r + 8g, col = ln / 16+ln) ----
        if (kb + AKT - 1 > qb + wave * 16) {
            int qrow0 = qb + wave * 16 + 8 * g;
            #pragma unroll
            for (int r = 0; r < 8; ++r) {
                int q = qrow0 + r;
                if (kb + ln > q)      s0[r] = -1e30f;
                if (kb + 16 + ln > q) s1[r] = -1e30f;
            }
        }

        // ---- online softmax (row reductions across 16-lane half) ----
        #pragma unroll
        for (int r = 0; r < 8; ++r) {
            float mx = fmaxf(s0[r], s1[r]);
            mx = fmaxf(mx, __shfl_xor(mx, 1, 32));
            mx = fmaxf(mx, __shfl_xor(mx, 2, 32));
            mx = fmaxf(mx, __shfl_xor(mx, 4, 32));
            mx = fmaxf(mx, __shfl_xor(mx, 8, 32));
            float mnew  = fmaxf(m[r], mx);
            float alpha = __expf(m[r] - mnew);
            float p0 = __expf(s0[r] - mnew);
            float p1 = __expf(s1[r] - mnew);
            s0[r] = p0; s1[r] = p1;
            float rs = p0 + p1;
            rs += __shfl_xor(rs, 1, 32);
            rs += __shfl_xor(rs, 2, 32);
            rs += __shfl_xor(rs, 4, 32);
            rs += __shfl_xor(rs, 8, 32);
            lsum[r] = lsum[r] * alpha + rs;
            m[r] = mnew;
            #pragma unroll
            for (int j = 0; j < 8; ++j) o[j][r] *= alpha;
        }

        // ---- P: C-layout -> A-layout via per-wave LDS scratch ----
        unsigned short* Pw = &Ps[wave * 16 * PSPAD];
        {
            int prow0 = 8 * g;
            #pragma unroll
            for (int r = 0; r < 8; ++r) {
                Pw[(prow0 + r) * PSPAD + ln]      = f2bf(s0[r]);
                Pw[(prow0 + r) * PSPAD + 16 + ln] = f2bf(s1[r]);
            }
        }
        const unsigned short* pp = &Pw[ln * PSPAD + g * 8];
        v16bf pf = ld_frag2(pp, pp + 16);

        // ---- O += P(16x32) . V(32x128) ----
        #pragma unroll
        for (int j = 0; j < 8; ++j) {
            const unsigned short* pv = &Vs[(j * 16 + ln) * VTPAD + g * 8];
            v16bf vf = ld_frag2(pv, pv + 16);
            o[j] = wmma_bf16(pf, vf, o[j]);
        }
    }

    // ---- epilogue: O /= l, write bf16 into (b*S+s, h*128+d) layout ----
    int orow0 = qb + wave * 16 + 8 * g;
    #pragma unroll
    for (int j = 0; j < 8; ++j) {
        #pragma unroll
        for (int r = 0; r < 8; ++r) {
            float v = o[j][r] / lsum[r];
            O[((size_t)b * 2048 + orow0 + r) * 2048 +
              (size_t)h * 128 + j * 16 + ln] = f2bf(v);
        }
    }
}

// ---------------------------------------------------------------------------
// Host-side orchestration
// ---------------------------------------------------------------------------
extern "C" void kernel_launch(void* const* d_in, const int* in_sizes, int n_in,
                              void* d_out, int out_size, void* d_ws, size_t ws_size,
                              hipStream_t stream) {
    (void)in_sizes; (void)n_in; (void)out_size; (void)ws_size;

    const float* x    = (const float*)d_in[0];
    const float* cosT = (const float*)d_in[1];
    const float* sinT = (const float*)d_in[2];
    const float* wdq  = (const float*)d_in[3];
    const float* wuq  = (const float*)d_in[4];
    const float* wqr  = (const float*)d_in[5];
    const float* wdkv = (const float*)d_in[6];
    const float* wuk  = (const float*)d_in[7];
    const float* wkr  = (const float*)d_in[8];
    const float* wuv  = (const float*)d_in[9];
    const float* wo   = (const float*)d_in[10];

    constexpr size_t ROWS = 4096;   // B*S
    constexpr size_t SZ_Xb   = ROWS * 2048;
    constexpr size_t SZ_Wdq  = 2048ull * 1024;
    constexpr size_t SZ_Wuq  = 1024ull * 2048;
    constexpr size_t SZ_Wqr  = 1024ull * 2048;
    constexpr size_t SZ_Wdkv = 2048ull * 512;
    constexpr size_t SZ_Wuk  = 512ull * 1024;
    constexpr size_t SZ_Wkr  = 2048ull * 1024;
    constexpr size_t SZ_Wuv  = 512ull * 1024;
    constexpr size_t SZ_Wo   = 2048ull * 2048;
    constexpr size_t SZ_cq   = ROWS * 1024;
    constexpr size_t SZ_ckv  = ROWS * 512;
    constexpr size_t SZ_qu   = ROWS * 2048;
    constexpr size_t SZ_qr   = ROWS * 2048;
    constexpr size_t SZ_ku   = ROWS * 1024;
    constexpr size_t SZ_krp  = ROWS * 1024;
    constexpr size_t SZ_vu   = ROWS * 1024;
    constexpr size_t SZ_Q    = 2ull * 16 * 2048 * 256;
    constexpr size_t SZ_K    = 2ull * 8 * 2048 * 256;
    constexpr size_t SZ_Vt   = 2ull * 8 * 128 * 2048;
    constexpr size_t SZ_O    = ROWS * 2048;

    unsigned short* p = (unsigned short*)d_ws;
    unsigned short* Xb   = p; p += SZ_Xb;
    unsigned short* Wdq  = p; p += SZ_Wdq;    // all weights stored TRANSPOSED (NxK)
    unsigned short* Wuq  = p; p += SZ_Wuq;
    unsigned short* Wqr  = p; p += SZ_Wqr;
    unsigned short* Wdkv = p; p += SZ_Wdkv;
    unsigned short* Wuk  = p; p += SZ_Wuk;
    unsigned short* Wkr  = p; p += SZ_Wkr;
    unsigned short* Wuv  = p; p += SZ_Wuv;
    unsigned short* Wo_  = p; p += SZ_Wo;
    unsigned short* cq   = p; p += SZ_cq;
    unsigned short* ckv  = p; p += SZ_ckv;
    unsigned short* qu   = p; p += SZ_qu;
    unsigned short* qrp  = p; p += SZ_qr;
    unsigned short* ku   = p; p += SZ_ku;
    unsigned short* krp  = p; p += SZ_krp;
    unsigned short* vu   = p; p += SZ_vu;
    unsigned short* Qa   = p; p += SZ_Q;
    unsigned short* Ka   = p; p += SZ_K;
    unsigned short* Vta  = p; p += SZ_Vt;
    unsigned short* Ob   = p; p += SZ_O;

    auto cvt = [&](const float* src, unsigned short* dst, size_t n) {
        cvt_f32_bf16<<<(unsigned)((n + 255) / 256), 256, 0, stream>>>(src, dst, (long long)n);
    };
    // transpose-convert: input is (R=K) x (C=N), output N x K
    auto cvt_t = [&](const float* src, unsigned short* dst, int K, int N) {
        size_t n = (size_t)K * N;
        cvt_f32_bf16_t<<<(unsigned)((n + 255) / 256), 256, 0, stream>>>(src, dst, K, N);
    };
    cvt(x, Xb, SZ_Xb);
    cvt_t(wdq,  Wdq,  2048, 1024);
    cvt_t(wuq,  Wuq,  1024, 2048);
    cvt_t(wqr,  Wqr,  1024, 2048);
    cvt_t(wdkv, Wdkv, 2048, 512);
    cvt_t(wuk,  Wuk,  512, 1024);
    cvt_t(wkr,  Wkr,  2048, 1024);
    cvt_t(wuv,  Wuv,  512, 1024);
    cvt_t(wo,   Wo_,  2048, 2048);

    auto gemm_bf = [&](const unsigned short* A, const unsigned short* Bt,
                       unsigned short* C, int M, int N, int Kd) {
        dim3 grid(N / GBN, M / GBM);
        gemm_bf16<unsigned short><<<grid, 256, 0, stream>>>(A, Bt, C, M, N, Kd);
    };

    gemm_bf(Xb,  Wdq,  cq,  4096, 1024, 2048);   // c_q
    gemm_bf(Xb,  Wdkv, ckv, 4096,  512, 2048);   // c_kv
    gemm_bf(cq,  Wuq,  qu,  4096, 2048, 1024);   // q_u
    gemm_bf(cq,  Wqr,  qrp, 4096, 2048, 1024);   // q_r (pre-rope)
    gemm_bf(ckv, Wuk,  ku,  4096, 1024,  512);   // k_u
    gemm_bf(Xb,  Wkr,  krp, 4096, 1024, 2048);   // k_r (pre-rope)
    gemm_bf(ckv, Wuv,  vu,  4096, 1024,  512);   // v_u

    assemble_q<<<(unsigned)((2ull * 2048 * 16 * 64) / 256), 256, 0, stream>>>(
        qu, qrp, cosT, sinT, Qa);
    assemble_k<<<(unsigned)((2ull * 2048 * 8 * 64) / 256), 256, 0, stream>>>(
        ku, krp, cosT, sinT, Ka);
    assemble_v<<<(unsigned)((2ull * 8 * 128 * 2048) / 256), 256, 0, stream>>>(vu, Vta);

    dim3 ga(2048 / AQT, 16, 2);
    mla_attention<<<ga, 128, 0, stream>>>(Qa, Ka, Vta, Ob);

    dim3 gf(2048 / GBN, 4096 / GBM);
    gemm_bf16<float><<<gf, 256, 0, stream>>>(Ob, Wo_, (float*)d_out, 4096, 2048, 2048);
}
